// SelfAttention_27496380629769
// MI455X (gfx1250) — matrix operand order repaired and checked by reading, not compiled, not run on hardware
//
#include <hip/hip_runtime.h>
#include <hip/hip_bf16.h>

// ---------------------------------------------------------------------------
// Self-attention for MI455X (gfx1250): bf16 WMMA flash-attention pipeline.
//   K1 proj_kernel   : per-head Q/K/V projection f32 -> bf16 (V stored [d][seq]);
//                      1/sqrt(EMBED) folded into the Q projection.
//   K2 attn_kernel   : flash attention, 8 waves = 8 heads, double-buffered
//                      additive mask-bias tile in LDS, lazy max-rescale.
//   K3 outproj_kernel: (N*SEQ,512) @ Wout^T + bout -> f32 out
// All matmuls use v_wmma_f32_16x16x32_bf16 (wave32, 16x16 tiles, K=32).
// ---------------------------------------------------------------------------

#define NBATCH 4
#define SEQLEN 2048
#define EMBED  512
#define NHEADS 8
#define HDIM   64

typedef __bf16 bf16_t;
typedef __attribute__((ext_vector_type(16))) __bf16    v16bf;
typedef __attribute__((ext_vector_type(8)))  __bf16    v8bf;
typedef __attribute__((ext_vector_type(8)))  float     v8f;
typedef __attribute__((ext_vector_type(4)))  float     v4f;
typedef __attribute__((ext_vector_type(4)))  unsigned  u32x4;

// native f32 -> bf16 convert (v_cvt_pk_bf16_f32 on gfx1250)
static __device__ __forceinline__ bf16_t f2bf(float f) { return (bf16_t)f; }
static __device__ __forceinline__ unsigned short f2bfbits(float f) {
    bf16_t b = (bf16_t)f;
    return __builtin_bit_cast(unsigned short, b);
}

// load 8 packed bf16 (16B, aligned) as v8bf
static __device__ __forceinline__ v8bf load8bf(const unsigned short* p) {
    u32x4 u = *(const u32x4*)p;
    return __builtin_bit_cast(v8bf, u);
}
static __device__ __forceinline__ void store8bf(unsigned short* p, v8bf v) {
    *(u32x4*)p = __builtin_bit_cast(u32x4, v);
}
// load 8 contiguous f32 (two b128) and convert to v8bf in one vector convert
static __device__ __forceinline__ v8bf cvt8(const float* p) {
    v4f a = *(const v4f*)p;
    v4f b = *(const v4f*)(p + 4);
    v8f x = __builtin_shufflevector(a, b, 0, 1, 2, 3, 4, 5, 6, 7);
    return __builtin_convertvector(x, v8bf);
}
static __device__ __forceinline__ v16bf cat16(v8bf lo, v8bf hi) {
    return __builtin_shufflevector(lo, hi, 0, 1, 2, 3, 4, 5, 6, 7,
                                            8, 9, 10, 11, 12, 13, 14, 15);
}
static __device__ __forceinline__ v16bf load16bf(const unsigned short* p) {
    return cat16(load8bf(p), load8bf(p + 8));
}
static __device__ __forceinline__ v16bf cvt16(const float* p) {
    return cat16(cvt8(p), cvt8(p + 8));
}
static __device__ __forceinline__ v8f wmma_bf16(v16bf a, v16bf b, v8f c) {
    // D = A(16x32) * B(32x16) + C, f32 accumulate
    return __builtin_amdgcn_wmma_f32_16x16x32_bf16(
        /*neg_a=*/false, a, /*neg_b=*/false, b,
        /*c_mod=*/(short)0, c, /*reuse_a=*/false, /*reuse_b=*/false);
}

// ---------------------------------------------------------------------------
// Kernel 1: per-head projection.  out[l][e] = oscale * sum_d x[l][d]*W[e][d].
// grid: (32, N*H) block: 128 (4 waves).  Each wave: 16 seq rows x 64 cols.
// transposed==0 -> out[nh][l][e] row-major ; transposed!=0 -> out[nh][e][l].
// ---------------------------------------------------------------------------
__global__ __launch_bounds__(128) void proj_kernel(
    const float* __restrict__ x, const float* __restrict__ W,
    unsigned short* __restrict__ out, int transposed, float oscale)
{
    const int lane  = threadIdx.x & 31;
    const int wave  = threadIdx.x >> 5;
    const int nh    = blockIdx.y;            // n*NHEADS + h
    const int n     = nh >> 3;
    const int h     = nh & 7;
    const int rowbase = (blockIdx.x * 4 + wave) * 16;
    const int r     = lane & 15;             // row (A) / column (B,C)
    const int khalf = lane >> 4;

    // A operand: 16 rows of x, K=64 in two 16x32 chunks (contiguous k order: k = 16*khalf + s)
    const float* xrow = x + ((size_t)n * SEQLEN + rowbase + r) * EMBED + h * HDIM;
    v16bf a[2];
#pragma unroll
    for (int c = 0; c < 2; ++c) {
        const float* p = xrow + c * 32 + khalf * 16;
        a[c] = cat16(cvt8(p), cvt8(p + 8));
    }

#pragma unroll
    for (int t = 0; t < 4; ++t) {            // 4 column tiles of 16
        const int e = t * 16 + r;            // output column handled by lane
        v8f acc = {};
#pragma unroll
        for (int c = 0; c < 2; ++c) {
            // B[k][n] = W[n][k]: row e of W, 16 contiguous f32 along k
            acc = wmma_bf16(a[c], cvt16(W + (size_t)e * HDIM + c * 32 + khalf * 16), acc);
        }
        if (!transposed) {
            unsigned short* o = out + ((size_t)nh * SEQLEN + rowbase) * HDIM;
#pragma unroll
            for (int j = 0; j < 8; ++j)
                o[(size_t)(khalf * 8 + j) * HDIM + e] = f2bfbits(acc[j] * oscale);
        } else {
            // V^T store: rows of C are contiguous along seq -> one 16B store
            v8bf pk;
#pragma unroll
            for (int j = 0; j < 8; ++j) pk[j] = f2bf(acc[j] * oscale);
            store8bf(out + ((size_t)nh * HDIM + e) * SEQLEN + rowbase + khalf * 8, pk);
        }
    }
}

// ---------------------------------------------------------------------------
// Kernel 2: flash attention.  grid: (SEQ/16, N), block: 256 (wave w == head w).
// Each wave owns 16 query rows for one head; loops over keys in blocks of 32.
// Mask bias tile (16x32 f32, 0 or -1e30) double-buffered in LDS, shared by
// all 8 heads; one barrier per iteration; lazy rescale on max update.
// ---------------------------------------------------------------------------
__global__ __launch_bounds__(256) void attn_kernel(
    const unsigned short* __restrict__ qg,   // [N*H][SEQ][64] bf16 (pre-scaled)
    const unsigned short* __restrict__ kg,   // [N*H][SEQ][64] bf16
    const unsigned short* __restrict__ vgT,  // [N*H][64][SEQ] bf16 (transposed)
    const int*            __restrict__ mask, // [N][1][SEQ][SEQ]
    unsigned short*       __restrict__ aout) // [N][SEQ][512] bf16
{
    __shared__ float sbias[2][16 * 32];
    __shared__ __align__(16) unsigned short sP[NHEADS][16 * 32];

    const int tid   = threadIdx.x;
    const int lane  = tid & 31;
    const int h     = tid >> 5;              // wave index == head
    const int n     = blockIdx.y;
    const int qbase = blockIdx.x * 16;
    const int r     = lane & 15;
    const int khalf = lane >> 4;
    const int nh    = n * NHEADS + h;
    const int row0  = khalf * 8;             // first C-layout row held by lane

    const int* mrow = mask + ((size_t)n * SEQLEN + qbase) * SEQLEN;
    // cooperative: convert a 16x32 mask tile into additive f32 bias in LDS
    auto loadBias = [&](int kb, int buf) {
        const int e  = tid * 2;
        const int mr = e >> 5, mc = e & 31;
        sbias[buf][e]     = (mrow[(size_t)mr * SEQLEN + kb + mc]     == 0) ? -1.0e30f : 0.0f;
        sbias[buf][e + 1] = (mrow[(size_t)mr * SEQLEN + kb + mc + 1] == 0) ? -1.0e30f : 0.0f;
    };

    // Q operand (A layout), kept in registers for the whole loop
    const unsigned short* qrow = qg + ((size_t)nh * SEQLEN + qbase + r) * HDIM;
    v16bf aq[2];
#pragma unroll
    for (int c = 0; c < 2; ++c) {
        const unsigned short* p = qrow + c * 32 + khalf * 16;
        aq[c] = cat16(load8bf(p), load8bf(p + 8));
    }

    v8f O[4] = {};                            // 16x64 f32 accumulator
    float m[8], l[8];
#pragma unroll
    for (int j = 0; j < 8; ++j) { m[j] = -3.0e30f; l[j] = 0.0f; }

    loadBias(0, 0);
    __syncthreads();

    for (int it = 0, kb0 = 0; kb0 < SEQLEN; kb0 += 32, ++it) {
        const int cur = it & 1;
        if (kb0 + 32 < SEQLEN) {
            loadBias(kb0 + 32, cur ^ 1);     // overlaps with compute below
            __builtin_prefetch(kg + ((size_t)nh * SEQLEN + kb0 + 32 + r) * HDIM, 0, 1);
        }

        // S = Q @ K^T : two 16x16 key tiles, K-dim = 64 (2 chunks of 32)
        v8f s0 = {}, s1 = {};
#pragma unroll
        for (int c = 0; c < 2; ++c) {
            const unsigned short* k0 =
                kg + ((size_t)nh * SEQLEN + kb0 + r) * HDIM + c * 32 + khalf * 16;
            s0 = wmma_bf16(aq[c], load16bf(k0), s0);
            const unsigned short* k1 =
                kg + ((size_t)nh * SEQLEN + kb0 + 16 + r) * HDIM + c * 32 + khalf * 16;
            s1 = wmma_bf16(aq[c], load16bf(k1), s1);
        }

        // add mask bias, find per-row block max (16-lane xor reductions)
        const float* bias = sbias[cur];
        float x0[8], x1[8], mnew[8];
        bool chg = false;
#pragma unroll
        for (int j = 0; j < 8; ++j) {
            const int mr = row0 + j;
            x0[j] = s0[j] + bias[mr * 32 + r];
            x1[j] = s1[j] + bias[mr * 32 + 16 + r];
            float t = fmaxf(x0[j], x1[j]);
#pragma unroll
            for (int d = 1; d < 16; d <<= 1) t = fmaxf(t, __shfl_xor(t, d, 32));
            mnew[j] = fmaxf(m[j], t);
            chg = chg || (mnew[j] > m[j]);
        }
        // lazy rescale: only when some row's max actually moved (wave-uniform)
        if (__any((int)chg)) {
#pragma unroll
            for (int j = 0; j < 8; ++j) {
                const float alpha = __expf(m[j] - mnew[j]);
                l[j] *= alpha;
                m[j]  = mnew[j];
#pragma unroll
                for (int t = 0; t < 4; ++t) O[t][j] *= alpha;
            }
        }
        // P = exp(x - m), row sums, and P -> LDS (bf16, C layout)
        unsigned short* Pw = sP[h];
#pragma unroll
        for (int j = 0; j < 8; ++j) {
            const float p0 = __expf(x0[j] - m[j]);
            const float p1 = __expf(x1[j] - m[j]);
            float su = p0 + p1;
#pragma unroll
            for (int d = 1; d < 16; d <<= 1) su += __shfl_xor(su, d, 32);
            l[j] += su;
            Pw[(row0 + j) * 32 + r]      = f2bfbits(p0);
            Pw[(row0 + j) * 32 + 16 + r] = f2bfbits(p1);
        }
        // read P back in A layout, O += P(16x32) @ V(32x64)
        const unsigned short* Pr = sP[h] + (size_t)r * 32 + khalf * 16;
        v16bf aP = cat16(load8bf(Pr), load8bf(Pr + 8));
#pragma unroll
        for (int t = 0; t < 4; ++t) {
            const unsigned short* vp =
                vgT + ((size_t)nh * HDIM + t * 16 + r) * SEQLEN + kb0 + khalf * 16;
            O[t] = wmma_bf16(aP, load16bf(vp), O[t]);
        }
        __syncthreads();                     // single barrier per iteration
    }

    // epilogue: O /= l (via reciprocal), store bf16 head-slice of concat output
    float linv[8];
#pragma unroll
    for (int j = 0; j < 8; ++j) linv[j] = 1.0f / l[j];
    unsigned short* orow = aout + ((size_t)n * SEQLEN + qbase) * EMBED + h * HDIM;
#pragma unroll
    for (int t = 0; t < 4; ++t) {
#pragma unroll
        for (int j = 0; j < 8; ++j) {
            orow[(size_t)(row0 + j) * EMBED + t * 16 + r] = f2bfbits(O[t][j] * linv[j]);
        }
    }
}

// ---------------------------------------------------------------------------
// Kernel 3: output projection.  out = attn @ Wout^T + bout (f32 out).
// grid: (N*SEQ/64, EMBED/64), block: 128.  Each wave: 16 rows x 64 cols, K=512.
// ---------------------------------------------------------------------------
__global__ __launch_bounds__(128) void outproj_kernel(
    const unsigned short* __restrict__ a,    // [N*SEQ][512] bf16
    const float* __restrict__ Wout,          // [512][512] f32
    const float* __restrict__ bout,          // [512]
    float* __restrict__ out)                 // [N*SEQ][512] f32
{
    const int lane    = threadIdx.x & 31;
    const int wave    = threadIdx.x >> 5;
    const int rowbase = (blockIdx.x * 4 + wave) * 16;
    const int colbase = blockIdx.y * 64;
    const int r       = lane & 15;
    const int khalf   = lane >> 4;

    const unsigned short* arow = a + (size_t)(rowbase + r) * EMBED;
    v8f acc[4] = {};
    for (int c = 0; c < EMBED; c += 32) {
        const unsigned short* ap = arow + c + khalf * 16;
        v16bf av = cat16(load8bf(ap), load8bf(ap + 8));
#pragma unroll
        for (int t = 0; t < 4; ++t) {
            const int e = colbase + t * 16 + r;
            acc[t] = wmma_bf16(av, cvt16(Wout + (size_t)e * EMBED + c + khalf * 16), acc[t]);
        }
    }
#pragma unroll
    for (int t = 0; t < 4; ++t) {
        const int e = colbase + t * 16 + r;
        const float bias = bout[e];
#pragma unroll
        for (int j = 0; j < 8; ++j)
            out[(size_t)(rowbase + khalf * 8 + j) * EMBED + e] = acc[t][j] + bias;
    }
}

// ---------------------------------------------------------------------------
extern "C" void kernel_launch(void* const* d_in, const int* in_sizes, int n_in,
                              void* d_out, int out_size, void* d_ws, size_t ws_size,
                              hipStream_t stream) {
    const float* values = (const float*)d_in[0];
    const float* keys   = (const float*)d_in[1];
    const float* query  = (const float*)d_in[2];
    const int*   mask   = (const int*)  d_in[3];
    const float* Wv     = (const float*)d_in[4];
    const float* Wk     = (const float*)d_in[5];
    const float* Wq     = (const float*)d_in[6];
    const float* Wout   = (const float*)d_in[7];
    const float* bout   = (const float*)d_in[8];
    float*       outp   = (float*)d_out;

    // workspace: qb | kb | vbT | attn_out, all bf16 (raw u16 bits). 32 MB total.
    const size_t QKV = (size_t)NBATCH * NHEADS * SEQLEN * HDIM;   // 4M elems
    unsigned short* qb   = (unsigned short*)d_ws;
    unsigned short* kb   = qb + QKV;
    unsigned short* vbT  = kb + QKV;
    unsigned short* aout = vbT + QKV;

    const float qscale = 0.044194173824159216f;   // 1/sqrt(EMBED) folded into Q
    const dim3 pjGrid(SEQLEN / 16 / 4, NBATCH * NHEADS);          // (32, 32)
    proj_kernel<<<pjGrid, 128, 0, stream>>>(query,  Wq, qb,  0, qscale);
    proj_kernel<<<pjGrid, 128, 0, stream>>>(keys,   Wk, kb,  0, 1.0f);
    proj_kernel<<<pjGrid, 128, 0, stream>>>(values, Wv, vbT, 1, 1.0f);

    attn_kernel<<<dim3(SEQLEN / 16, NBATCH), 256, 0, stream>>>(qb, kb, vbT, mask, aout);

    outproj_kernel<<<dim3(NBATCH * SEQLEN / 64, EMBED / 64), 128, 0, stream>>>(
        aout, Wout, bout, outp);
}
